// MixtureOfExperts_9156870275389
// MI455X (gfx1250) — matrix operand order repaired and checked by reading, not compiled, not run on hardware
//
#include <hip/hip_runtime.h>
#include <math.h>

// ---------------- problem dims (match reference) ----------------
#define T_TOK   8192      // B*S = 4*2048
#define DMODEL  1024
#define DFF     4096
#define NEXP    8

// ---------------- GEMM tiling ----------------
#define BM 128
#define BN 128
#define BK 32             // K-step of v_wmma_f32_16x16x32_bf16

typedef __bf16 bf16_t;
typedef bf16_t v16bf __attribute__((ext_vector_type(16)));
typedef float  v8f   __attribute__((ext_vector_type(8)));

union FragAB {
  v16bf v;
  uint4 q[2];
};

__device__ __forceinline__ unsigned short f2bf(float f) {
  unsigned int u = __float_as_uint(f);
  u += 0x7FFFu + ((u >> 16) & 1u);          // round-to-nearest-even
  return (unsigned short)(u >> 16);
}

__device__ __forceinline__ float gelu_tanh(float v) {
  float t = tanhf(0.7978845608028654f * (v + 0.044715f * v * v * v));
  return 0.5f * v * (1.0f + t);
}

// ---------------- fp32 -> bf16 conversion of x (row-major kept) ----------------
__global__ __launch_bounds__(256) void cvt_x_kernel(const float* __restrict__ x,
                                                    unsigned short* __restrict__ xbf) {
  int i = blockIdx.x * 256 + threadIdx.x;           // handles 4 elements
  float4 v = ((const float4*)x)[i];
  uint2 o;
  o.x = (unsigned int)f2bf(v.x) | ((unsigned int)f2bf(v.y) << 16);
  o.y = (unsigned int)f2bf(v.z) | ((unsigned int)f2bf(v.w) << 16);
  ((uint2*)xbf)[i] = o;
}

// ---------------- fp32 -> bf16 convert + transpose of weights ----------------
// src: [K][N] fp32 (per expert, blockIdx.z); dst: [N][K] bf16. 32x32 tiles.
__global__ __launch_bounds__(256) void cvt_w_kernel(const float* __restrict__ src,
                                                    unsigned short* __restrict__ dst,
                                                    int K, int N) {
  __shared__ unsigned short t[32][34];              // +2 pad: dodge bank conflicts
  const size_t eoff = (size_t)blockIdx.z * K * N;
  src += eoff;
  dst += eoff;
  const int k0 = blockIdx.y * 32, n0 = blockIdx.x * 32;
  const int tid = threadIdx.x;
#pragma unroll
  for (int i = 0; i < 4; ++i) {
    int idx = i * 256 + tid;
    int k = idx >> 5, n = idx & 31;                 // coalesced along n
    t[n][k] = f2bf(src[(size_t)(k0 + k) * N + n0 + n]);
  }
  __syncthreads();
#pragma unroll
  for (int i = 0; i < 2; ++i) {
    int idx = i * 256 + tid;
    int n = idx >> 4, kp = (idx & 15) << 1;         // coalesced along k
    unsigned int u = (unsigned int)t[n][kp] | ((unsigned int)t[n][kp + 1] << 16);
    *(unsigned int*)&dst[(size_t)(n0 + n) * K + k0 + kp] = u;
  }
}

// ---------------- gating: softmax + top-2 weights ----------------
// one wave32 per token; gw[t][e] = prob[e] if e in top-2 else 0
__global__ __launch_bounds__(256) void gate_kernel(const float* __restrict__ x,
                                                   const float* __restrict__ Wg,
                                                   const float* __restrict__ bg,
                                                   float* __restrict__ gw) {
  int tok  = (blockIdx.x * 256 + threadIdx.x) >> 5;
  int lane = threadIdx.x & 31;
  const float* xr = x + (size_t)tok * DMODEL;
  float acc[NEXP];
#pragma unroll
  for (int e = 0; e < NEXP; ++e) acc[e] = 0.0f;
  for (int d = lane; d < DMODEL; d += 32) {
    float xv = xr[d];
    const float* wr = Wg + (size_t)d * NEXP;
#pragma unroll
    for (int e = 0; e < NEXP; ++e) acc[e] = fmaf(xv, wr[e], acc[e]);
  }
#pragma unroll
  for (int off = 16; off > 0; off >>= 1)
#pragma unroll
    for (int e = 0; e < NEXP; ++e) acc[e] += __shfl_down(acc[e], off, 32);
  if (lane == 0) {
    float l[NEXP], p[NEXP];
    float mx = -1e30f;
#pragma unroll
    for (int e = 0; e < NEXP; ++e) { l[e] = acc[e] + bg[e]; mx = fmaxf(mx, l[e]); }
    float s = 0.0f;
#pragma unroll
    for (int e = 0; e < NEXP; ++e) { p[e] = __expf(l[e] - mx); s += p[e]; }
    float inv = 1.0f / s;
#pragma unroll
    for (int e = 0; e < NEXP; ++e) p[e] *= inv;
    int i1 = 0;
#pragma unroll
    for (int e = 1; e < NEXP; ++e) if (p[e] > p[i1]) i1 = e;     // first max wins ties
    int i2 = (i1 == 0) ? 1 : 0;
#pragma unroll
    for (int e = 0; e < NEXP; ++e) if (e != i1 && p[e] > p[i2]) i2 = e;
    float* g = gw + (size_t)tok * NEXP;
#pragma unroll
    for (int e = 0; e < NEXP; ++e) g[e] = (e == i1 || e == i2) ? p[e] : 0.0f;
  }
}

// ---------------- WMMA mainloop (shared by both GEMMs) ----------------
// A:  bf16 row-major [T][KDIM]   (rows = output M)
// BT: bf16 row-major [N][KDIM]   (rows = output N; pre-transposed weights)
// 256 threads = 8 waves; wave (wm,wn) in 2x4 grid owns 64x32; 4x2 WMMA tiles.
// Software pipeline: register-staged b128 copies overlap WMMA of current panel.
template <int KDIM>
__device__ __forceinline__ void gemm_mainloop(const unsigned short* __restrict__ A,
                                              const unsigned short* __restrict__ BT,
                                              int m0, int n0, v8f acc[4][2]) {
  __shared__ __align__(16) unsigned short At[BM][BK];   // [m][k]
  __shared__ __align__(16) unsigned short Bt[BN][BK];   // [n][k]

  const int tid  = threadIdx.x;
  const int w    = tid >> 5;
  const int lane = tid & 31;
  const int wm = w & 1, wn = w >> 1;
  const int hl = lane >> 4, l16 = lane & 15;

  // each thread copies 32 contiguous bytes of each tile: 2 x b128
  const int rowc = tid >> 1;                 // 0..127
  const int kc   = (tid & 1) << 4;           // 0 or 16 (halfs)
  const unsigned short* gA = A  + (size_t)(m0 + rowc) * KDIM + kc;
  const unsigned short* gB = BT + (size_t)(n0 + rowc) * KDIM + kc;

  uint4 ra0 = *(const uint4*)(gA);
  uint4 ra1 = *(const uint4*)(gA + 8);
  uint4 rb0 = *(const uint4*)(gB);
  uint4 rb1 = *(const uint4*)(gB + 8);

  for (int k0 = 0; k0 < KDIM; k0 += BK) {
    // LDS is free here (barrier at end of previous iteration)
    *(uint4*)&At[rowc][kc]     = ra0;
    *(uint4*)&At[rowc][kc + 8] = ra1;
    *(uint4*)&Bt[rowc][kc]     = rb0;
    *(uint4*)&Bt[rowc][kc + 8] = rb1;

    // issue next panel's global loads (hidden behind this panel's WMMAs)
    if (k0 + BK < KDIM) {
      const unsigned short* nA = gA + k0 + BK;
      const unsigned short* nB = gB + k0 + BK;
      ra0 = *(const uint4*)(nA);
      ra1 = *(const uint4*)(nA + 8);
      rb0 = *(const uint4*)(nB);
      rb1 = *(const uint4*)(nB + 8);
      // warm L2 two panels ahead (global_prefetch_b8)
      if (k0 + 2 * BK < KDIM) {
        __builtin_prefetch(gA + k0 + 2 * BK, 0, 0);
        __builtin_prefetch(gB + k0 + 2 * BK, 0, 0);
      }
    }
    __syncthreads();   // staged panel visible

    // --- fragments per ISA 16-bit layouts ---
    FragAB a[4], b[2];
#pragma unroll
    for (int mt = 0; mt < 4; ++mt) {
      int r  = wm * 64 + mt * 16 + l16;     // matrix row M
      int kk = hl * 8;                      // lanes 0-15: K 0..7 & 16..23; 16-31: 8..15 & 24..31
      a[mt].q[0] = *(const uint4*)&At[r][kk];
      a[mt].q[1] = *(const uint4*)&At[r][kk + 16];
    }
#pragma unroll
    for (int nt = 0; nt < 2; ++nt) {
      int n  = wn * 32 + nt * 16 + l16;     // matrix col N
      int kk = hl * 8;                      // lanes 0-15: K 0..7 & 16..23; 16-31: 8..15 & 24..31
      b[nt].q[0] = *(const uint4*)&Bt[n][kk];
      b[nt].q[1] = *(const uint4*)&Bt[n][kk + 16];
    }
#pragma unroll
    for (int mt = 0; mt < 4; ++mt)
#pragma unroll
      for (int nt = 0; nt < 2; ++nt)
        acc[mt][nt] = __builtin_amdgcn_wmma_f32_16x16x32_bf16(
            false, a[mt].v, false, b[nt].v, (short)0, acc[mt][nt], false, false);
    __syncthreads();   // fragment reads done; LDS reusable
  }
}

// ---------------- expert FFN layer 1: h = gelu(x @ W1 + b1), bf16 out ----------------
__global__ __launch_bounds__(256) void ffn1_kernel(const unsigned short* __restrict__ xbf,
                                                   const unsigned short* __restrict__ W1T,
                                                   const float* __restrict__ b1,
                                                   unsigned short* __restrict__ h) {
  const int m0 = blockIdx.y * BM;
  const int n0 = blockIdx.x * BN;
  v8f acc[4][2];
#pragma unroll
  for (int mt = 0; mt < 4; ++mt)
#pragma unroll
    for (int nt = 0; nt < 2; ++nt)
#pragma unroll
      for (int r = 0; r < 8; ++r) acc[mt][nt][r] = 0.0f;

  gemm_mainloop<DMODEL>(xbf, W1T, m0, n0, acc);

  const int lane = threadIdx.x & 31;
  const int w = threadIdx.x >> 5;
  const int wm = w & 1, wn = w >> 1;
  const int hl = lane >> 4, l16 = lane & 15;
#pragma unroll
  for (int mt = 0; mt < 4; ++mt)
#pragma unroll
    for (int nt = 0; nt < 2; ++nt) {
      int n = n0 + wn * 32 + nt * 16 + l16;
      float bn = b1[n];
#pragma unroll
      for (int r = 0; r < 8; ++r) {
        int m = m0 + wm * 64 + mt * 16 + hl * 8 + r;
        float v = acc[mt][nt][r] + bn;
        h[(size_t)m * DFF + n] = f2bf(gelu_tanh(v));
      }
    }
}

// ---------------- expert FFN layer 2: out (+)= gw * (h @ W2 + b2) ----------------
__global__ __launch_bounds__(256) void ffn2_kernel(const unsigned short* __restrict__ h,
                                                   const unsigned short* __restrict__ W2T,
                                                   const float* __restrict__ b2,
                                                   const float* __restrict__ gw,
                                                   int expert, float* __restrict__ out) {
  const int m0 = blockIdx.y * BM;
  const int n0 = blockIdx.x * BN;
  v8f acc[4][2];
#pragma unroll
  for (int mt = 0; mt < 4; ++mt)
#pragma unroll
    for (int nt = 0; nt < 2; ++nt)
#pragma unroll
      for (int r = 0; r < 8; ++r) acc[mt][nt][r] = 0.0f;

  gemm_mainloop<DFF>(h, W2T, m0, n0, acc);

  const int lane = threadIdx.x & 31;
  const int w = threadIdx.x >> 5;
  const int wm = w & 1, wn = w >> 1;
  const int hl = lane >> 4, l16 = lane & 15;
#pragma unroll
  for (int mt = 0; mt < 4; ++mt) {
    float wrow[8];
#pragma unroll
    for (int r = 0; r < 8; ++r) {
      int m = m0 + wm * 64 + mt * 16 + hl * 8 + r;
      wrow[r] = gw[(size_t)m * NEXP + expert];
    }
#pragma unroll
    for (int nt = 0; nt < 2; ++nt) {
      int n = n0 + wn * 32 + nt * 16 + l16;
      float bn = b2[n];
#pragma unroll
      for (int r = 0; r < 8; ++r) {
        int m = m0 + wm * 64 + mt * 16 + hl * 8 + r;
        size_t o = (size_t)m * DMODEL + n;
        float v = wrow[r] * (acc[mt][nt][r] + bn);
        out[o] = (expert == 0) ? v : (out[o] + v);   // expert 0 initializes d_out
      }
    }
  }
}

// ---------------- launcher ----------------
extern "C" void kernel_launch(void* const* d_in, const int* in_sizes, int n_in,
                              void* d_out, int out_size, void* d_ws, size_t ws_size,
                              hipStream_t stream) {
  const float* x  = (const float*)d_in[0];
  const float* Wg = (const float*)d_in[1];
  const float* bg = (const float*)d_in[2];
  const float* W1 = (const float*)d_in[3];
  const float* b1 = (const float*)d_in[4];
  const float* W2 = (const float*)d_in[5];
  const float* b2 = (const float*)d_in[6];
  float* out = (float*)d_out;

  // ws layout: gw (256 KB) | x_bf16 (16 MB) | h_bf16 (64 MB) | W1T (64 MB) | W2T (64 MB)
  char* ws = (char*)d_ws;
  const size_t GW_BYTES  = (size_t)T_TOK * NEXP * sizeof(float);
  const size_t XBF_BYTES = (size_t)T_TOK * DMODEL * sizeof(unsigned short);
  const size_t H_BYTES   = (size_t)T_TOK * DFF * sizeof(unsigned short);
  const size_t W_BYTES   = (size_t)NEXP * DMODEL * DFF * sizeof(unsigned short);
  float* gw           = (float*)ws;
  unsigned short* xbf = (unsigned short*)(ws + GW_BYTES);
  unsigned short* h   = (unsigned short*)(ws + GW_BYTES + XBF_BYTES);
  unsigned short* w1t = (unsigned short*)(ws + GW_BYTES + XBF_BYTES + H_BYTES);
  unsigned short* w2t = (unsigned short*)(ws + GW_BYTES + XBF_BYTES + H_BYTES + W_BYTES);

  cvt_x_kernel<<<(T_TOK * DMODEL) / (256 * 4), 256, 0, stream>>>(x, xbf);
  gate_kernel<<<T_TOK / 8, 256, 0, stream>>>(x, Wg, bg, gw);   // 8 waves (tokens) per block
  // W1 [E][D][F] -> W1T [E][F][D];  W2 [E][F][D] -> W2T [E][D][F]
  cvt_w_kernel<<<dim3(DFF / 32, DMODEL / 32, NEXP), 256, 0, stream>>>(W1, w1t, DMODEL, DFF);
  cvt_w_kernel<<<dim3(DMODEL / 32, DFF / 32, NEXP), 256, 0, stream>>>(W2, w2t, DFF, DMODEL);

  for (int e = 0; e < NEXP; ++e) {
    ffn1_kernel<<<dim3(DFF / BN, T_TOK / BM), 256, 0, stream>>>(
        xbf, w1t + (size_t)e * DFF * DMODEL, b1 + (size_t)e * DFF, h);
    ffn2_kernel<<<dim3(DMODEL / BN, T_TOK / BM), 256, 0, stream>>>(
        h, w2t + (size_t)e * DMODEL * DFF, b2 + (size_t)e * DMODEL, gw, e, out);
  }
}
